// CycleGNNLayer_9509057593729
// MI455X (gfx1250) — matrix operand order repaired and checked by reading, not compiled, not run on hardware
//
#include <hip/hip_runtime.h>
#include <hip/hip_bf16.h>

// CDNA5 (gfx1250, wave32) GNN layer: GRU message -> PNA aggregate -> LSTM edge
// update -> LayerNorm, all matmuls on v_wmma_f32_16x16x32_bf16.
// Elementwise transcendentals use raw v_rcp/v_exp (and v_tanh if available),
// bf16 packing uses native converts, LN reductions use ds_swizzle.

#define N_NODES 50000
#define N_EDGES 400000
#define DIM     64

typedef __attribute__((ext_vector_type(16))) __bf16          v16bf;
typedef __attribute__((ext_vector_type(16))) unsigned short  v16u;
typedef __attribute__((ext_vector_type(8)))  float           v8f;

// ---------- fast math helpers ----------
__device__ __forceinline__ float fast_rcp(float x) {
#if __has_builtin(__builtin_amdgcn_rcpf)
  return __builtin_amdgcn_rcpf(x);
#else
  return 1.0f / x;
#endif
}
__device__ __forceinline__ float fast_rsq(float x) {
#if __has_builtin(__builtin_amdgcn_rsqf)
  return __builtin_amdgcn_rsqf(x);
#else
  return rsqrtf(x);
#endif
}
__device__ __forceinline__ float sigmf(float x) {
  return fast_rcp(1.0f + __expf(-x));
}
__device__ __forceinline__ float fast_tanh(float x) {
#if __has_builtin(__builtin_amdgcn_tanhf)
  return __builtin_amdgcn_tanhf(x);
#else
  // tanh(x) = 2*sigmoid(2x) - 1 ; saturates correctly at +-1
  float t = fast_rcp(1.0f + __expf(-2.0f * x));
  return __builtin_fmaf(2.0f, t, -1.0f);
#endif
}

__device__ __forceinline__ v8f wmma_bf16(v16bf a, v16bf b, v8f c) {
  return __builtin_amdgcn_wmma_f32_16x16x32_bf16(false, a, false, b, (short)0, c,
                                                 false, false);
}

// B fragment (32x16, bf16) pre-swizzled in ws: one 32B vector per lane.
__device__ __forceinline__ v16bf load_b_frag(const unsigned short* __restrict__ frags,
                                             int fid, int lane) {
  const v16u* p = (const v16u*)(frags + ((size_t)fid * 32 + lane) * 16);
  return __builtin_bit_cast(v16bf, *p);
}

// A fragment (16x32, bf16) built from one f32 row (lane holds row lane%16).
// element i<8 : K = kk*32 + 8*hi + i ; element i>=8 : K = kk*32 + 16 + 8*hi + (i-8)
__device__ __forceinline__ v16bf make_a_frag(const float* __restrict__ row, int kk,
                                             int hi) {
  v16bf t;
  int b0 = kk * 32 + 8 * hi;
  int b1 = b0 + 16;
#pragma unroll
  for (int i = 0; i < 8; ++i) t[i] = (__bf16)row[b0 + i];
#pragma unroll
  for (int i = 0; i < 8; ++i) t[8 + i] = (__bf16)row[b1 + i];
  return t;
}

__device__ __forceinline__ v16bf make_a_frag_add(const float* __restrict__ ra,
                                                 const float* __restrict__ rb,
                                                 int kk, int hi) {
  v16bf t;
  int b0 = kk * 32 + 8 * hi;
  int b1 = b0 + 16;
#pragma unroll
  for (int i = 0; i < 8; ++i) t[i] = (__bf16)(ra[b0 + i] + rb[b0 + i]);
#pragma unroll
  for (int i = 0; i < 8; ++i) t[8 + i] = (__bf16)(ra[b1 + i] + rb[b1 + i]);
  return t;
}

__device__ __forceinline__ void atomicMaxF(float* p, float v) {
  if (v >= 0.0f) atomicMax((int*)p, __float_as_int(v));
  else           atomicMin((unsigned int*)p, __float_as_uint(v));
}
__device__ __forceinline__ void atomicMinF(float* p, float v) {
  if (v >= 0.0f) atomicMin((int*)p, __float_as_int(v));
  else           atomicMax((unsigned int*)p, __float_as_uint(v));
}

// ds_swizzle XOR-lane add (group-of-32 mode: and=0x1f, or=0, xor=M).
template <int M>
__device__ __forceinline__ float xor_add(float v) {
  int o = __builtin_amdgcn_ds_swizzle(__float_as_int(v), (M << 10) | 0x1f);
  return v + __int_as_float(o);
}

// Row LayerNorm over 64 cols held as val[jt][rr] per lane; reduce across the
// 16-lane half (xor masks 1,2,4,8 stay within a half on wave32).
__device__ __forceinline__ void ln_write(const float val[4][8], int base_row, int r,
                                         int hi, const float* __restrict__ ln_g,
                                         const float* __restrict__ ln_b,
                                         float* __restrict__ out) {
#pragma unroll
  for (int rr = 0; rr < 8; ++rr) {
    float s = 0.0f, s2 = 0.0f;
#pragma unroll
    for (int j = 0; j < 4; ++j) { float v = val[j][rr]; s += v; s2 += v * v; }
    s = xor_add<1>(s);  s = xor_add<2>(s);  s = xor_add<4>(s);  s = xor_add<8>(s);
    s2 = xor_add<1>(s2); s2 = xor_add<2>(s2); s2 = xor_add<4>(s2); s2 = xor_add<8>(s2);
    float mean = s * (1.0f / 64.0f);
    float var  = fmaxf(s2 * (1.0f / 64.0f) - mean * mean, 0.0f);
    float rstd = fast_rsq(var + 1e-5f);
    int row = base_row + rr + 8 * hi;
#pragma unroll
    for (int j = 0; j < 4; ++j) {
      int d = j * 16 + r;
      out[(size_t)row * 64 + d] = (val[j][rr] - mean) * rstd * ln_g[d] + ln_b[d];
    }
  }
}

// ---------- kernel 0: init stats ----------
__global__ void init_kernel(float* __restrict__ sum, float* __restrict__ sumsq,
                            float* __restrict__ maxb, float* __restrict__ minb,
                            float* __restrict__ deg_in, float* __restrict__ deg_out,
                            float* __restrict__ avgd) {
  int i = blockIdx.x * blockDim.x + threadIdx.x;
  if (i < N_NODES * DIM) {
    sum[i]   = 0.0f;
    sumsq[i] = 0.0f;
    maxb[i]  = __int_as_float(0xFF800000);  // -inf
    minb[i]  = __int_as_float(0x7F800000);  // +inf
  }
  if (i < N_NODES) { deg_in[i] = 0.0f; deg_out[i] = 0.0f; }
  if (i == 0) avgd[0] = 0.0f;
}

// ---------- kernel 1: swizzle weights into WMMA B-fragment layout ----------
// frag fid: [0,24) gru_Wih(j,kk) | [24,48) gru_Whh | [48,80) lstm_Wih |
//           [80,112) lstm_Whh | [112,144) pna_W (j in 0..3, kk in 0..7)
__global__ void prep_weights_kernel(const float* __restrict__ gWih,
                                    const float* __restrict__ gWhh,
                                    const float* __restrict__ lWih,
                                    const float* __restrict__ lWhh,
                                    const float* __restrict__ pW,
                                    unsigned short* __restrict__ frags) {
  int tid = blockIdx.x * blockDim.x + threadIdx.x;
  if (tid >= 144 * 32) return;
  int fid = tid >> 5, lane = tid & 31;
  const float* W; int f, j, kk, ldk;
  if (fid < 24)       { W = gWih; f = fid;       j = f >> 1; kk = f & 1; ldk = 64; }
  else if (fid < 48)  { W = gWhh; f = fid - 24;  j = f >> 1; kk = f & 1; ldk = 64; }
  else if (fid < 80)  { W = lWih; f = fid - 48;  j = f >> 1; kk = f & 1; ldk = 64; }
  else if (fid < 112) { W = lWhh; f = fid - 80;  j = f >> 1; kk = f & 1; ldk = 64; }
  else                { W = pW;   f = fid - 112; j = f >> 3; kk = f & 7; ldk = 256; }
  int row = j * 16 + (lane & 15);
  int col = kk * 32 + 8 * (lane >> 4);
  unsigned short* o = frags + ((size_t)fid * 32 + lane) * 16;
#pragma unroll
  for (int i = 0; i < 16; ++i)
    o[i] = __builtin_bit_cast(unsigned short, (__bf16)W[(size_t)row * ldk + col + (i < 8 ? i : 8 + i)]);
}

// ---------- kernel 2: degrees ----------
__global__ void degree_kernel(const int* __restrict__ src, const int* __restrict__ dst,
                              float* __restrict__ deg_in, float* __restrict__ deg_out) {
  int e = blockIdx.x * blockDim.x + threadIdx.x;
  if (e >= N_EDGES) return;
  atomicAdd(deg_in + dst[e], 1.0f);
  atomicAdd(deg_out + src[e], 1.0f);
}

// ---------- kernel 3: sum of log(deg_out+1) ----------
__global__ void avgd_kernel(const float* __restrict__ deg_out, float* __restrict__ avgd) {
  int i = blockIdx.x * blockDim.x + threadIdx.x;
  float v = (i < N_NODES) ? logf(deg_out[i] + 1.0f) : 0.0f;
  v = xor_add<1>(v); v = xor_add<2>(v); v = xor_add<4>(v);
  v = xor_add<8>(v); v = xor_add<16>(v);
  if ((threadIdx.x & 31) == 0) atomicAdd(avgd, v);
}

// ---------- kernel 4: GRU messages + PNA scatter (WMMA) ----------
__global__ __launch_bounds__(256) void gru_edge_kernel(
    const int* __restrict__ src, const int* __restrict__ dst,
    const int* __restrict__ etype, const float* __restrict__ efeat,
    const float* __restrict__ nfeat, const float* __restrict__ rel_emb,
    const unsigned short* __restrict__ frags, const float* __restrict__ bih,
    const float* __restrict__ bhh, float* __restrict__ sum,
    float* __restrict__ sumsq, float* __restrict__ maxb, float* __restrict__ minb) {
  int wave = (blockIdx.x * blockDim.x + threadIdx.x) >> 5;  // 25000 waves exactly
  int lane = threadIdx.x & 31;
  int r = lane & 15, hi = lane >> 4;
  int e0 = wave * 16;
  int ea = e0 + r;

  const float* ex = efeat   + (size_t)ea * 64;
  const float* rl = rel_emb + (size_t)etype[ea] * 64;
  const float* hx = nfeat   + (size_t)src[ea] * 64;
  v16bf ax[2], ah[2];
#pragma unroll
  for (int kk = 0; kk < 2; ++kk) {
    ax[kk] = make_a_frag_add(ex, rl, kk, hi);
    ah[kk] = make_a_frag(hx, kk, hi);
  }
  int srcv[8], dstv[8];
#pragma unroll
  for (int rr = 0; rr < 8; ++rr) {
    int e = e0 + rr + 8 * hi;
    srcv[rr] = src[e];
    dstv[rr] = dst[e];
  }

#pragma unroll
  for (int jt = 0; jt < 4; ++jt) {
    // r gate: cols [jt*16, +16): gi + gh fused accumulate
    v8f cr = {};
    cr = wmma_bf16(ax[0], load_b_frag(frags, jt * 2 + 0, lane), cr);
    cr = wmma_bf16(ax[1], load_b_frag(frags, jt * 2 + 1, lane), cr);
    cr = wmma_bf16(ah[0], load_b_frag(frags, 24 + jt * 2 + 0, lane), cr);
    cr = wmma_bf16(ah[1], load_b_frag(frags, 24 + jt * 2 + 1, lane), cr);
    // z gate: tile jt+4
    v8f cz = {};
    cz = wmma_bf16(ax[0], load_b_frag(frags, (jt + 4) * 2 + 0, lane), cz);
    cz = wmma_bf16(ax[1], load_b_frag(frags, (jt + 4) * 2 + 1, lane), cz);
    cz = wmma_bf16(ah[0], load_b_frag(frags, 24 + (jt + 4) * 2 + 0, lane), cz);
    cz = wmma_bf16(ah[1], load_b_frag(frags, 24 + (jt + 4) * 2 + 1, lane), cz);
    // n gate needs i_n and h_n separately (tile jt+8)
    v8f cin = {};
    cin = wmma_bf16(ax[0], load_b_frag(frags, (jt + 8) * 2 + 0, lane), cin);
    cin = wmma_bf16(ax[1], load_b_frag(frags, (jt + 8) * 2 + 1, lane), cin);
    v8f chn = {};
    chn = wmma_bf16(ah[0], load_b_frag(frags, 24 + (jt + 8) * 2 + 0, lane), chn);
    chn = wmma_bf16(ah[1], load_b_frag(frags, 24 + (jt + 8) * 2 + 1, lane), chn);

    float br  = bih[jt * 16 + r] + bhh[jt * 16 + r];
    float bz  = bih[(jt + 4) * 16 + r] + bhh[(jt + 4) * 16 + r];
    float bni = bih[(jt + 8) * 16 + r];
    float bnh = bhh[(jt + 8) * 16 + r];
    int d = jt * 16 + r;
#pragma unroll
    for (int rr = 0; rr < 8; ++rr) {
      float rg = sigmf(cr[rr] + br);
      float zg = sigmf(cz[rr] + bz);
      float ng = fast_tanh(cin[rr] + bni + rg * (chn[rr] + bnh));
      float hv = nfeat[(size_t)srcv[rr] * 64 + d];
      float m  = (1.0f - zg) * ng + zg * hv;
      size_t a = (size_t)dstv[rr] * 64 + d;
      atomicAdd(sum + a, m);
      atomicAdd(sumsq + a, m * m);
      atomicMaxF(maxb + a, m);
      atomicMinF(minb + a, m);
    }
  }
}

// ---------- kernel 5: PNA node update (WMMA) + LN ----------
__global__ __launch_bounds__(256) void pna_node_kernel(
    const float* __restrict__ sum, const float* __restrict__ sumsq,
    const float* __restrict__ maxb, const float* __restrict__ minb,
    const float* __restrict__ deg_in, const float* __restrict__ avgd,
    const unsigned short* __restrict__ frags, const float* __restrict__ pna_b,
    const float* __restrict__ ln_g, const float* __restrict__ ln_b,
    float* __restrict__ nfeat_new, float* __restrict__ out1) {
  int wave = (blockIdx.x * blockDim.x + threadIdx.x) >> 5;
  if (wave >= N_NODES / 16) return;
  int lane = threadIdx.x & 31;
  int r = lane & 15, hi = lane >> 4;
  int n0 = wave * 16;
  int node_a = n0 + r;

  float deg_a = deg_in[node_a];
  float rden  = fast_rcp(fmaxf(deg_a, 1.0f));
  bool  has   = deg_a > 0.0f;
  const float* sa = sum   + (size_t)node_a * 64;
  const float* qa = sumsq + (size_t)node_a * 64;
  const float* xa = maxb  + (size_t)node_a * 64;
  const float* na = minb  + (size_t)node_a * 64;

  // agg row: [mean | max | min | std], K = 256 -> 8 fragments
  v16bf a[8];
#pragma unroll
  for (int kk = 0; kk < 8; ++kk) {
    int stat = kk >> 1;
    v16bf t;
#pragma unroll
    for (int i = 0; i < 16; ++i) {
      int K = kk * 32 + ((i < 8) ? (8 * hi + i) : (16 + 8 * hi + (i - 8)));
      int d = K & 63;
      float v;
      if (stat == 0)      v = sa[d] * rden;
      else if (stat == 1) v = has ? xa[d] : 0.0f;
      else if (stat == 2) v = has ? na[d] : 0.0f;
      else {
        float mu = sa[d] * rden;
        float q  = qa[d] * rden;
        v = sqrtf(fmaxf(q - mu * mu, 0.0f) + 1e-5f);
      }
      t[i] = (__bf16)v;
    }
    a[kk] = t;
  }

  v8f acc[4];
#pragma unroll
  for (int j = 0; j < 4; ++j) {
    v8f c = {};
#pragma unroll
    for (int kk = 0; kk < 8; ++kk)
      c = wmma_bf16(a[kk], load_b_frag(frags, 112 + j * 8 + kk, lane), c);
    acc[j] = c;
  }

  float inv_avg_d = (float)N_NODES * fast_rcp(avgd[0]);
  float val[4][8];
#pragma unroll
  for (int rr = 0; rr < 8; ++rr) {
    int node = n0 + rr + 8 * hi;
    float scale = logf(deg_in[node] + 1.0f) * inv_avg_d;
#pragma unroll
    for (int j = 0; j < 4; ++j) {
      float v = (acc[j][rr] + pna_b[j * 16 + r]) * scale;
      val[j][rr] = v;
      nfeat_new[(size_t)node * 64 + j * 16 + r] = v;
    }
  }
  ln_write(val, n0, r, hi, ln_g, ln_b, out1);
}

// ---------- kernel 6: LSTM edge update (WMMA) + LN ----------
__global__ __launch_bounds__(256) void lstm_edge_kernel(
    const int* __restrict__ dst, const float* __restrict__ efeat,
    const float* __restrict__ equery, const float* __restrict__ nfeat_new,
    const unsigned short* __restrict__ frags, const float* __restrict__ bih,
    const float* __restrict__ bhh, const float* __restrict__ ln_g,
    const float* __restrict__ ln_b, float* __restrict__ out0,
    float* __restrict__ out2) {
  int wave = (blockIdx.x * blockDim.x + threadIdx.x) >> 5;  // 25000 waves exactly
  int lane = threadIdx.x & 31;
  int r = lane & 15, hi = lane >> 4;
  int e0 = wave * 16;
  int ea = e0 + r;

  const float* nf = nfeat_new + (size_t)dst[ea] * 64;
  const float* ef = efeat + (size_t)ea * 64;
  v16bf an[2], ae[2];
#pragma unroll
  for (int kk = 0; kk < 2; ++kk) {
    an[kk] = make_a_frag(nf, kk, hi);
    ae[kk] = make_a_frag(ef, kk, hi);
  }

  float hn[4][8], cn[4][8];
#pragma unroll
  for (int jt = 0; jt < 4; ++jt) {
    v8f cg4[4];  // gate tiles i,f,g,o at j = jt, jt+4, jt+8, jt+12
    float bias[4];
#pragma unroll
    for (int g = 0; g < 4; ++g) {
      int j = jt + 4 * g;
      v8f c = {};
      c = wmma_bf16(an[0], load_b_frag(frags, 48 + j * 2 + 0, lane), c);
      c = wmma_bf16(an[1], load_b_frag(frags, 48 + j * 2 + 1, lane), c);
      c = wmma_bf16(ae[0], load_b_frag(frags, 80 + j * 2 + 0, lane), c);
      c = wmma_bf16(ae[1], load_b_frag(frags, 80 + j * 2 + 1, lane), c);
      cg4[g] = c;
      bias[g] = bih[j * 16 + r] + bhh[j * 16 + r];
    }
    int d = jt * 16 + r;
#pragma unroll
    for (int rr = 0; rr < 8; ++rr) {
      int e = e0 + rr + 8 * hi;
      float ii = sigmf(cg4[0][rr] + bias[0]);
      float ff = sigmf(cg4[1][rr] + bias[1]);
      float gg = fast_tanh(cg4[2][rr] + bias[2]);
      float oo = sigmf(cg4[3][rr] + bias[3]);
      float cq = equery[(size_t)e * 64 + d];
      float c_ = ff * cq + ii * gg;
      cn[jt][rr] = c_;
      hn[jt][rr] = oo * fast_tanh(c_);
    }
  }
  ln_write(hn, e0, r, hi, ln_g, ln_b, out0);
  ln_write(cn, e0, r, hi, ln_g, ln_b, out2);
}

// ---------- launch ----------
extern "C" void kernel_launch(void* const* d_in, const int* in_sizes, int n_in,
                              void* d_out, int out_size, void* d_ws, size_t ws_size,
                              hipStream_t stream) {
  const int*   src      = (const int*)d_in[0];
  const int*   dst      = (const int*)d_in[1];
  const int*   etype    = (const int*)d_in[2];
  const float* efeat    = (const float*)d_in[3];
  const float* nfeat    = (const float*)d_in[4];
  const float* equery   = (const float*)d_in[5];
  const float* rel_emb  = (const float*)d_in[6];
  const float* gru_Wih  = (const float*)d_in[7];
  const float* gru_Whh  = (const float*)d_in[8];
  const float* gru_bih  = (const float*)d_in[9];
  const float* gru_bhh  = (const float*)d_in[10];
  const float* pna_W    = (const float*)d_in[11];
  const float* pna_b    = (const float*)d_in[12];
  const float* lstm_Wih = (const float*)d_in[13];
  const float* lstm_Whh = (const float*)d_in[14];
  const float* lstm_bih = (const float*)d_in[15];
  const float* lstm_bhh = (const float*)d_in[16];
  const float* ln_g     = (const float*)d_in[17];
  const float* ln_b     = (const float*)d_in[18];

  float* out  = (float*)d_out;
  float* out0 = out;                                   // LN(h_new)     [E,64]
  float* out1 = out0 + (size_t)N_EDGES * 64;           // LN(nfeat_new) [N,64]
  float* out2 = out1 + (size_t)N_NODES * 64;           // LN(c_new)     [E,64]

  float* ws        = (float*)d_ws;
  float* sum       = ws;                               // N*64
  float* sumsq     = sum + (size_t)N_NODES * 64;
  float* maxb      = sumsq + (size_t)N_NODES * 64;
  float* minb      = maxb + (size_t)N_NODES * 64;
  float* nfeat_new = minb + (size_t)N_NODES * 64;
  float* deg_in    = nfeat_new + (size_t)N_NODES * 64;
  float* deg_out   = deg_in + N_NODES;
  float* avgd      = deg_out + N_NODES;
  unsigned short* frags = (unsigned short*)(avgd + 8); // 144 frags * 512 u16

  hipLaunchKernelGGL(init_kernel, dim3((N_NODES * DIM + 255) / 256), dim3(256), 0,
                     stream, sum, sumsq, maxb, minb, deg_in, deg_out, avgd);
  hipLaunchKernelGGL(prep_weights_kernel, dim3((144 * 32 + 255) / 256), dim3(256), 0,
                     stream, gru_Wih, gru_Whh, lstm_Wih, lstm_Whh, pna_W, frags);
  hipLaunchKernelGGL(degree_kernel, dim3((N_EDGES + 255) / 256), dim3(256), 0,
                     stream, src, dst, deg_in, deg_out);
  hipLaunchKernelGGL(avgd_kernel, dim3((N_NODES + 255) / 256), dim3(256), 0, stream,
                     deg_out, avgd);
  // 25000 edge tiles = 3125 blocks * 8 waves (exact)
  hipLaunchKernelGGL(gru_edge_kernel, dim3(N_EDGES / 16 / 8), dim3(256), 0, stream,
                     src, dst, etype, efeat, nfeat, rel_emb, frags, gru_bih,
                     gru_bhh, sum, sumsq, maxb, minb);
  // 3125 node tiles, 8 waves/block -> 391 blocks (guarded)
  hipLaunchKernelGGL(pna_node_kernel, dim3((N_NODES / 16 + 7) / 8), dim3(256), 0,
                     stream, sum, sumsq, maxb, minb, deg_in, avgd, frags, pna_b,
                     ln_g, ln_b, nfeat_new, out1);
  hipLaunchKernelGGL(lstm_edge_kernel, dim3(N_EDGES / 16 / 8), dim3(256), 0, stream,
                     dst, efeat, equery, nfeat_new, frags, lstm_bih, lstm_bhh,
                     ln_g, ln_b, out0, out2);
}
